// SNN_8770323219204
// MI455X (gfx1250) — compile-verified, matrix operand order checked by reading, not run on hardware
//
#include <hip/hip_runtime.h>
#include <hip/hip_bf16.h>

// CDNA5 (gfx1250) wave32 WMMA fragment types
typedef __bf16  v16bf __attribute__((ext_vector_type(16)));
typedef __bf16  v8bf  __attribute__((ext_vector_type(8)));
typedef float   v8f   __attribute__((ext_vector_type(8)));

#define TH_HID   0.1f
#define TH_LAST  1.0e5f
#define ALPHA_V  0.99f

// ---------------------------------------------------------------------------
// Prep: f32 weights -> zero-padded bf16 [Mp, Kp] row-major.
// ---------------------------------------------------------------------------
__global__ void pad_weight_bf16(const float* __restrict__ W, __bf16* __restrict__ Wp,
                                int M, int K, int Kp, int total /* Mp*Kp */) {
  const int idx = blockIdx.x * blockDim.x + threadIdx.x;
  if (idx >= total) return;
  const int m = idx / Kp;
  const int k = idx - m * Kp;
  const float v = (m < M && k < K) ? W[(size_t)m * K + k] : 0.0f;
  Wp[idx] = (__bf16)v;
}

// ---------------------------------------------------------------------------
// Prep: f32 spikes [B, K, T] -> bf16 transposed+padded [B, Tp, Kp].
// Coalesced writes; pads written as zero (fully deterministic buffer).
// ---------------------------------------------------------------------------
__global__ void transpose_pad_spike(const float* __restrict__ X, __bf16* __restrict__ Xt,
                                    int K, int T, int Kp, int Tp, int total /* B*Tp*Kp */) {
  const int idx = blockIdx.x * blockDim.x + threadIdx.x;
  if (idx >= total) return;
  const int b = idx / (Tp * Kp);
  const int r = idx - b * (Tp * Kp);
  const int t = r / Kp;
  const int k = r - t * Kp;
  const float v = (t < T && k < K) ? X[((size_t)b * K + k) * T + t] : 0.0f;
  Xt[idx] = (__bf16)v;
}

// ---------------------------------------------------------------------------
// Batched GEMM via V_WMMA_F32_16X16X32_BF16 (branch-free inner loop).
//   Z[b, m, n] = sum_k Wp[m,k] * Xt[b, n, k]
// One wave per 16x16 tile. Fragment layouts per ISA 7.12.2:
//   lane half (lane>>4) selects K offset +8; A.h[0] = K 0..7, A.h[1] = K 16..23.
//   B is the lane-symmetric mirror (lane&15 = column n), so with Xt storing
//   k-contiguous rows per timestep, both fragments are 2x global_load_b128.
// ---------------------------------------------------------------------------
template <int KP>
__global__ void gemm_wmma_bf16(const __bf16* __restrict__ Wp,  // [Mp, KP]
                               const __bf16* __restrict__ Xt,  // [B, Tp, KP]
                               float*        __restrict__ Z,   // [B, M, N]
                               int M, int N, int Tp) {
  const int lane = threadIdx.x & 31;
  const int half = lane >> 4;   // K offset +8 within fragment
  const int mn   = lane & 15;   // A row / B col / D col
  const int n0 = blockIdx.x * 16;
  const int m0 = blockIdx.y * 16;
  const int b  = blockIdx.z;

  const __bf16* __restrict__ aRow = Wp + (size_t)(m0 + mn) * KP;
  const __bf16* __restrict__ bRow = Xt + ((size_t)b * Tp + (n0 + mn)) * KP;
  const int kh = 8 * half;

  v8f acc = {};
#pragma unroll 2
  for (int k0 = 0; k0 < KP; k0 += 32) {
    union { v16bf v; v8bf h[2]; } A, Bv;
    A.h[0]  = *(const v8bf*)(aRow + k0 + kh);        // K = k0+0..7  (+8*half)
    A.h[1]  = *(const v8bf*)(aRow + k0 + 16 + kh);   // K = k0+16..23(+8*half)
    Bv.h[0] = *(const v8bf*)(bRow + k0 + kh);
    Bv.h[1] = *(const v8bf*)(bRow + k0 + 16 + kh);
    acc = __builtin_amdgcn_wmma_f32_16x16x32_bf16(
        /*neg_a=*/false, A.v, /*neg_b=*/false, Bv.v,
        /*c_mod=*/(short)0, acc, /*reuse_a=*/false, /*reuse_b=*/false);
  }

  float* __restrict__ Zb = Z + (size_t)b * M * N;
#pragma unroll
  for (int r = 0; r < 8; ++r) {
    const int m = m0 + r + 8 * half;
    const int n = n0 + mn;
    if (m < M && n < N) Zb[(size_t)m * N + n] = acc[r];
  }
}

// ---------------------------------------------------------------------------
// CUBA-LIF scan (alpha_c = 0 collapses the current state):
//   v_t = 0.99*v_{t-1} + z_t ; spike = (v >= th) ; v *= (1 - spike)
// One thread per (b, o); emits bf16 spikes TRANSPOSED into [B, Tp, M]
// (k-contiguous per timestep) so the next GEMM's B-fragments load wide.
// ---------------------------------------------------------------------------
__global__ void lif_scan_spike_t(const float* __restrict__ Z,   // [B, M, T]
                                 __bf16*      __restrict__ St,  // [B, Tp, M]
                                 int M, int T, int Tp, int rows /* B*M */) {
  const int id = blockIdx.x * blockDim.x + threadIdx.x;
  if (id >= rows) return;
  const int b = id / M;
  const int o = id - b * M;
  const float* __restrict__ z = Z + (size_t)id * T;
  __bf16* __restrict__ s = St + (size_t)b * Tp * M + o;
  float v = 0.0f;
  for (int t = 0; t < T; ++t) {
    v = ALPHA_V * v + z[t];
    const float sp = (v >= TH_HID) ? 1.0f : 0.0f;
    s[(size_t)t * M] = (__bf16)sp;     // coalesced across o
    v *= (1.0f - sp);
  }
}

// Readout layer: emit the voltage trace (TH_LAST effectively never fires).
__global__ void lif_scan_voltage(const float* __restrict__ Z,  // [rows, T]
                                 float*       __restrict__ V,  // [rows, T]
                                 int T, int rows) {
  const int id = blockIdx.x * blockDim.x + threadIdx.x;
  if (id >= rows) return;
  const float* __restrict__ z = Z + (size_t)id * T;
  float*       __restrict__ o = V + (size_t)id * T;
  float v = 0.0f;
  for (int t = 0; t < T; ++t) {
    v = ALPHA_V * v + z[t];
    o[t] = v;                                   // out = v (spiking=False)
    const float sp = (v >= TH_LAST) ? 1.0f : 0.0f;
    v *= (1.0f - sp);
  }
}

// ---------------------------------------------------------------------------
extern "C" void kernel_launch(void* const* d_in, const int* in_sizes, int n_in,
                              void* d_out, int out_size, void* d_ws, size_t ws_size,
                              hipStream_t stream) {
  (void)in_sizes; (void)n_in; (void)out_size; (void)ws_size;

  const float* spike = (const float*)d_in[0];  // [32, 2312, 300]
  const float* W1    = (const float*)d_in[1];  // [64, 2312]
  const float* W2    = (const float*)d_in[2];  // [512, 64]
  const float* W3    = (const float*)d_in[3];  // [512, 512]
  const float* W4    = (const float*)d_in[4];  // [10, 512]
  float*       out   = (float*)d_out;          // [32, 10, 300]

  constexpr int B = 32, T = 300, Tp = 304;     // Tp = ceil16(300)
  constexpr int K1 = 2312, K1p = 2336;         // K1p = ceil32(2312)

  // Workspace carve-up (256B aligned blocks, identical every call).
  char* ws = (char*)d_ws;
  size_t off = 0;
  auto carve = [&](size_t bytes) {
    char* p = ws + off;
    off += (bytes + 255) & ~(size_t)255;
    return p;
  };
  __bf16* W1p = (__bf16*)carve((size_t)64  * K1p * 2);   // [64, 2336]
  __bf16* W2p = (__bf16*)carve((size_t)512 * 64  * 2);   // [512, 64]
  __bf16* W3p = (__bf16*)carve((size_t)512 * 512 * 2);   // [512, 512]
  __bf16* W4p = (__bf16*)carve((size_t)16  * 512 * 2);   // [16, 512]
  __bf16* X1t = (__bf16*)carve((size_t)B * Tp * K1p * 2);// [32, 304, 2336]
  float*  Zb  = (float*) carve((size_t)B * 512 * T * 4); // [32, 512, 300]
  __bf16* SA  = (__bf16*)carve((size_t)B * Tp * 512 * 2);// [32, 304, <=512]
  __bf16* SB  = (__bf16*)carve((size_t)B * Tp * 512 * 2);

  const dim3 wave(32);
  auto grid1d = [](long n) { return dim3((unsigned)((n + 255) / 256)); };

  // --- Prep: pad weights, transpose+pad spikes to fragment-friendly layouts.
  pad_weight_bf16<<<grid1d((long)64 * K1p), 256, 0, stream>>>(W1, W1p, 64, K1, K1p, 64 * K1p);
  pad_weight_bf16<<<grid1d((long)512 * 64), 256, 0, stream>>>(W2, W2p, 512, 64, 64, 512 * 64);
  pad_weight_bf16<<<grid1d((long)512 * 512), 256, 0, stream>>>(W3, W3p, 512, 512, 512, 512 * 512);
  pad_weight_bf16<<<grid1d((long)16 * 512), 256, 0, stream>>>(W4, W4p, 10, 512, 512, 16 * 512);
  transpose_pad_spike<<<grid1d((long)B * Tp * K1p), 256, 0, stream>>>(
      spike, X1t, K1, T, K1p, Tp, B * Tp * K1p);

  const int nT = Tp / 16;  // 19 column tiles

  // Layer 1: [64 x 2312] (dominant GEMM) -> Z ; scan -> SA [B, Tp, 64]
  gemm_wmma_bf16<K1p><<<dim3(nT, 4, B), wave, 0, stream>>>(W1p, X1t, Zb, 64, T, Tp);
  lif_scan_spike_t<<<grid1d(B * 64), 256, 0, stream>>>(Zb, SA, 64, T, Tp, B * 64);

  // Layer 2: [512 x 64] -> Z ; scan -> SB [B, Tp, 512]
  gemm_wmma_bf16<64><<<dim3(nT, 32, B), wave, 0, stream>>>(W2p, SA, Zb, 512, T, Tp);
  lif_scan_spike_t<<<grid1d(B * 512), 256, 0, stream>>>(Zb, SB, 512, T, Tp, B * 512);

  // Layer 3: [512 x 512] -> Z ; scan -> SA (ping-pong)
  gemm_wmma_bf16<512><<<dim3(nT, 32, B), wave, 0, stream>>>(W3p, SB, Zb, 512, T, Tp);
  lif_scan_spike_t<<<grid1d(B * 512), 256, 0, stream>>>(Zb, SA, 512, T, Tp, B * 512);

  // Layer 4 (readout, M=10 in one padded 16-tile): -> Z ; voltage scan -> out
  gemm_wmma_bf16<512><<<dim3(nT, 1, B), wave, 0, stream>>>(W4p, SA, Zb, 10, T, Tp);
  lif_scan_voltage<<<grid1d(B * 10), 256, 0, stream>>>(Zb, out, T, B * 10);
}
